// MultiHeadAttention_44092134261443
// MI455X (gfx1250) — compile-verified, matrix operand order checked by reading, not run on hardware
//
#include <hip/hip_runtime.h>
#include <hip/hip_bf16.h>

// ---------------- problem constants ----------------
constexpr int B = 4, S = 2048, D = 768, H = 8, DK = 96;
constexpr float RSCALE = 0.10206207261596577f;   // 1/sqrt(96)
constexpr float MASKVAL = -1000000000.0f;

// ---------------- WMMA types ----------------
typedef __attribute__((ext_vector_type(16))) __bf16 v16bf;
typedef __attribute__((ext_vector_type(2)))  __bf16 v2bf;
typedef __attribute__((ext_vector_type(8)))  float  v8f;

// Hardware f32->bf16 conversion (gfx1250 has native bf16 cvt); the packed
// variant collapses two conversions into one VALU op when the builtin exists.
__device__ __forceinline__ unsigned short f32_to_bf16(float f) {
  __bf16 h = (__bf16)f;
  return __builtin_bit_cast(unsigned short, h);
}
__device__ __forceinline__ unsigned pack_bf16x2(float lo, float hi) {
#if __has_builtin(__builtin_amdgcn_cvt_pk_bf16_f32)
  v2bf p = __builtin_amdgcn_cvt_pk_bf16_f32(lo, hi);
  return __builtin_bit_cast(unsigned, p);
#else
  v2bf p = { (__bf16)lo, (__bf16)hi };
  return __builtin_bit_cast(unsigned, p);
#endif
}

// K index held by VGPR pair `vv` for lane-half `half` (ISA 7.12.2, 16-bit A 16x32)
__device__ __forceinline__ int kmap(int vv, int half) {
  return ((vv & 3) * 2) + ((vv >> 2) * 16) + half * 8;
}

// Fragment from a row-major 16x32 bf16 tile t[idx*ld + k] where idx = lane&15.
// Serves A (idx=m) and B staged N-major/B^T (idx=n). Each lane's 16 bf16 are
// two contiguous 8-element runs -> two b128 loads. Requires: t 16B-aligned,
// ld % 8 == 0.
__device__ __forceinline__ v16bf load_row_frag(const unsigned short* t, int ld) {
  const int lane = threadIdx.x & 31, half = lane >> 4, idx = lane & 15;
  union { v16bf v; uint4 q[2]; } f;
  const unsigned short* base = t + idx * ld + 8 * half;
  f.q[0] = *(const uint4*)(base);        // k = 8h .. 8h+7
  f.q[1] = *(const uint4*)(base + 16);   // k = 16+8h .. 16+8h+7
  return f.v;
}
// Fragment from K-major storage t[k*ld + n] (strided fallback, used once/step)
__device__ __forceinline__ v16bf load_kmajor_frag(const unsigned short* t, int ld) {
  const int lane = threadIdx.x & 31, half = lane >> 4, n = lane & 15;
  union { v16bf v; unsigned short u[16]; } f;
#pragma unroll
  for (int vv = 0; vv < 8; ++vv) {
    int k = kmap(vv, half);
    f.u[2 * vv]     = t[k * ld + n];
    f.u[2 * vv + 1] = t[(k + 1) * ld + n];
  }
  return f.v;
}

#define WMMA_BF16(A, Bf, C) \
  __builtin_amdgcn_wmma_f32_16x16x32_bf16(false, (A), false, (Bf), (short)0, (C), false, false)

// =====================================================================
// Kernel 1: out[m,n] = x[m,:] @ W[n,:] + b[n]   (f32 in, bf16 out)
// block = 128 thr (4 waves), tile 64x64, K-step 32. B staged as B^T (N-major).
// =====================================================================
__global__ __launch_bounds__(128)
void proj_kernel(const float* __restrict__ x, const float* __restrict__ W,
                 const float* __restrict__ bias, unsigned short* __restrict__ out) {
  alignas(16) __shared__ unsigned short As[64 * 40];   // [m][k], ld 40
  alignas(16) __shared__ unsigned short Bt[64 * 40];   // [n][k], ld 40
  unsigned* As32 = (unsigned*)As;
  unsigned* Bt32 = (unsigned*)Bt;
  const int tid = threadIdx.x, lane = tid & 31, wave = tid >> 5;
  const int half = lane >> 4, nl = lane & 15;
  const int row0 = blockIdx.x * 64, n0 = blockIdx.y * 64;

  v8f c[4];
#pragma unroll
  for (int t = 0; t < 4; ++t) c[t] = (v8f){0.f,0.f,0.f,0.f,0.f,0.f,0.f,0.f};

  for (int k0 = 0; k0 < D; k0 += 32) {
    __syncthreads();
#pragma unroll
    for (int e = 0; e < 8; ++e) {                  // 64x32 f32 -> packed bf16x2
      int idx = e * 128 + tid, r = idx >> 4, p2 = idx & 15;
      float2 xv = *(const float2*)&x[(size_t)(row0 + r) * D + k0 + p2 * 2];
      As32[r * 20 + p2] = pack_bf16x2(xv.x, xv.y);
    }
#pragma unroll
    for (int e = 0; e < 8; ++e) {                  // Bt[n][k] = W[n0+n][k0+k]
      int idx = e * 128 + tid, n = idx >> 4, p2 = idx & 15;
      float2 wv = *(const float2*)&W[(size_t)(n0 + n) * D + k0 + p2 * 2];
      Bt32[n * 20 + p2] = pack_bf16x2(wv.x, wv.y);
    }
    __syncthreads();
    v16bf a = load_row_frag(&As[wave * 16 * 40], 40);
#pragma unroll
    for (int t = 0; t < 4; ++t)
      c[t] = WMMA_BF16(a, load_row_frag(&Bt[t * 16 * 40], 40), c[t]);
  }
#pragma unroll
  for (int t = 0; t < 4; ++t) {
    int gc = n0 + t * 16 + nl;
    float bb = bias[gc];
#pragma unroll
    for (int r = 0; r < 8; ++r) {
      int gr = row0 + wave * 16 + r + 8 * half;
      out[(size_t)gr * D + gc] = f32_to_bf16(c[t][r] + bb);
    }
  }
}

// =====================================================================
// Kernel 2: per-column (over i) online max/sum of masked scaled q.k^T
// grid (S/128, B*H); block 256 (8 waves); wave owns 16 columns j.
// kv B-fragments loaded once straight from global via b128 (rows N-major).
// =====================================================================
__global__ __launch_bounds__(256)
void colstats_kernel(const unsigned short* __restrict__ q,
                     const unsigned short* __restrict__ kv,
                     const unsigned char* __restrict__ amask,
                     float* __restrict__ colmax, float* __restrict__ rcolsum) {
  alignas(16) __shared__ unsigned short qs[16 * 104];  // ld 104
  __shared__ float padf[16];
  unsigned* qs32 = (unsigned*)qs;
  const int tid = threadIdx.x, lane = tid & 31, wave = tid >> 5;
  const int half = lane >> 4, nl = lane & 15;
  const int bh = blockIdx.y, b = bh / H, h = bh % H;
  const int jt0 = blockIdx.x * 128 + wave * 16;
  const unsigned short* qg  = q  + (size_t)b * S * D + h * DK;
  const unsigned short* kvg = kv + (size_t)b * S * D + h * DK;
  const unsigned* qg32 = (const unsigned*)qg;

  v16bf bkv[3];                                    // wave's 16 kv columns, in regs
#pragma unroll
  for (int kk = 0; kk < 3; ++kk)
    bkv[kk] = load_row_frag(kvg + (size_t)jt0 * D + kk * 32, D);

  float m = -INFINITY, s = 0.f;
  for (int i0 = 0; i0 < S; i0 += 16) {
    __syncthreads();
#pragma unroll
    for (int e = 0; e < 3; ++e) {                  // stage 16x96 q rows as b32
      int idx = e * 256 + tid, r = idx / 48, cc = idx % 48;
      qs32[r * 52 + cc] = qg32[(size_t)(i0 + r) * (D / 2) + cc];
    }
    if (tid < 16) padf[tid] = amask[(size_t)b * S + i0 + tid] ? 1.f : 0.f;
    __syncthreads();

    v8f acc = (v8f){0.f,0.f,0.f,0.f,0.f,0.f,0.f,0.f};
#pragma unroll
    for (int kk = 0; kk < 3; ++kk)
      acc = WMMA_BF16(load_row_frag(&qs[kk * 32], 104), bkv[kk], acc);

    const int j = jt0 + nl;
    float wv[8], tmax = -INFINITY;
#pragma unroll
    for (int r = 0; r < 8; ++r) {
      int il = r + 8 * half, i = i0 + il;
      float w = acc[r] * RSCALE;
      if (j > i || padf[il] != 0.f) w = MASKVAL;
      wv[r] = w; tmax = fmaxf(tmax, w);
    }
    tmax = fmaxf(tmax, __shfl_xor(tmax, 16, 32));
    float mn = fmaxf(m, tmax), su = 0.f;
#pragma unroll
    for (int r = 0; r < 8; ++r) su += __expf(wv[r] - mn);
    su += __shfl_xor(su, 16, 32);
    s = s * __expf(m - mn) + su;
    m = mn;
  }
  if (lane < 16) {
    colmax [(size_t)bh * S + jt0 + lane] = m;
    rcolsum[(size_t)bh * S + jt0 + lane] = 1.f / s;
  }
}

// =====================================================================
// Kernel 3: out[i,:] = sum_j exp(w'-colmax[j])*rcolsum[j] * v[j,:] (+x)
// grid (S/32, B*H); block 192 (6 waves). waves 0-3: p-tile; all: p@v.
// kv tiles double-buffered via global_load_async_to_lds_b128 (ASYNCcnt).
// =====================================================================
__global__ __launch_bounds__(192)
void attnout_kernel(const unsigned short* __restrict__ q,
                    const unsigned short* __restrict__ kv,
                    const unsigned char* __restrict__ amask,
                    const float* __restrict__ colmax, const float* __restrict__ rcolsum,
                    const float* __restrict__ x, float* __restrict__ attn) {
  alignas(16) __shared__ unsigned short qs[32 * 104];      // q rows, ld 104
  alignas(16) __shared__ unsigned short kvs[2][32 * 104];  // double-buffered kv
  alignas(16) __shared__ unsigned short ps[32 * 40];       // p tile, ld 40
  __shared__ float padf[32];
  unsigned* qs32 = (unsigned*)qs;
  const int tid = threadIdx.x, lane = tid & 31, wave = tid >> 5;
  const int half = lane >> 4, nl = lane & 15;
  const int bh = blockIdx.y, b = bh / H, h = bh % H;
  const int i0 = blockIdx.x * 32;
  const unsigned short* qg  = q  + (size_t)b * S * D + h * DK;
  const unsigned short* kvg = kv + (size_t)b * S * D + h * DK;
  const unsigned* qg32 = (const unsigned*)qg;

  // CDNA5 async copy: 32 rows x 192B -> LDS, 16B chunks, bypassing VGPRs.
  auto issue_kv_async = [&](int buf, int jbase) {
    const char* gb = (const char*)kvg + (size_t)jbase * (D * 2);
    char* lb = (char*)kvs[buf];
    for (int c = tid; c < 384; c += 192) {
      int row = c / 12, sub = c % 12;
      unsigned la = (unsigned)(unsigned long long)(lb + row * 208 + sub * 16);
      const char* ga = gb + (size_t)row * (D * 2) + sub * 16;
      asm volatile("global_load_async_to_lds_b128 %0, %1, off"
                   :: "v"(la), "v"(ga) : "memory");
    }
  };

#pragma unroll
  for (int e = 0; e < 8; ++e) {                    // stage 32x96 q rows as b32
    int idx = e * 192 + tid, r = idx / 48, cc = idx % 48;
    qs32[r * 52 + cc] = qg32[(size_t)(i0 + r) * (D / 2) + cc];
  }
  if (tid < 32) padf[tid] = amask[(size_t)b * S + i0 + tid] ? 1.f : 0.f;
  issue_kv_async(0, 0);                            // prime the pipeline
  __syncthreads();

  const int rh = wave >> 1, ch = wave & 1;         // phase-1 roles (waves 0-3)
  v16bf aq[3];
  if (wave < 4) {
#pragma unroll
    for (int kk = 0; kk < 3; ++kk)
      aq[kk] = load_row_frag(&qs[rh * 16 * 104 + kk * 32], 104);
  }
  v8f oacc[2];
  oacc[0] = (v8f){0.f,0.f,0.f,0.f,0.f,0.f,0.f,0.f};
  oacc[1] = oacc[0];

  for (int j0 = 0; j0 < S; j0 += 32) {
    const int cur = (j0 >> 5) & 1;
    // previous iteration's consumers are done (loop-end barrier), so the
    // other buffer is free: issue its async fill, then retire only the two
    // OLDER async ops (this buffer) -- s_wait_asynccnt 2 keeps the new pair
    // in flight while we compute.
    if (j0 + 32 < S) {
      issue_kv_async(cur ^ 1, j0 + 32);
      asm volatile("s_wait_asynccnt 0x2" ::: "memory");
    } else {
      asm volatile("s_wait_asynccnt 0x0" ::: "memory");
    }
    __syncthreads();                               // all waves' fills visible

    if (wave < 4) {
      const int jc = j0 + ch * 16 + nl;
      const float cm = colmax[(size_t)bh * S + jc];
      const float rs = rcolsum[(size_t)bh * S + jc];
      v8f wacc = (v8f){0.f,0.f,0.f,0.f,0.f,0.f,0.f,0.f};
#pragma unroll
      for (int kk = 0; kk < 3; ++kk)
        wacc = WMMA_BF16(aq[kk],
                         load_row_frag(&kvs[cur][(ch * 16) * 104 + kk * 32], 104),
                         wacc);
#pragma unroll
      for (int r = 0; r < 8; ++r) {
        int il = rh * 16 + r + 8 * half, i = i0 + il;
        float w = wacc[r] * RSCALE;
        if (jc > i || padf[il] != 0.f) w = MASKVAL;
        ps[il * 40 + ch * 16 + nl] = f32_to_bf16(__expf(w - cm) * rs);
      }
    }
    __syncthreads();

    v16bf bv = load_kmajor_frag(&kvs[cur][wave * 16], 104);  // B[j][dk-col]
#pragma unroll
    for (int r2 = 0; r2 < 2; ++r2)
      oacc[r2] = WMMA_BF16(load_row_frag(&ps[r2 * 16 * 40], 40), bv, oacc[r2]);
    __syncthreads();                               // frees kvs[cur] and ps
  }

#pragma unroll
  for (int r2 = 0; r2 < 2; ++r2)
#pragma unroll
    for (int r = 0; r < 8; ++r) {
      int gi = i0 + r2 * 16 + r + 8 * half;
      int gc = h * DK + wave * 16 + nl;
      size_t off = ((size_t)b * S + gi) * D + gc;
      attn[off] = oacc[r2][r] + x[off];            // residual fused here
    }
}

// =====================================================================
// Kernel 4: LayerNorm over D=768; one block per row
// =====================================================================
__global__ __launch_bounds__(256)
void ln_kernel(const float* __restrict__ attn, const float* __restrict__ gamma,
               const float* __restrict__ beta, float* __restrict__ out) {
  const int tid = threadIdx.x, lane = tid & 31, wave = tid >> 5;
  const size_t row = blockIdx.x;
  float v[3], s = 0.f, s2 = 0.f;
#pragma unroll
  for (int e = 0; e < 3; ++e) {
    v[e] = attn[row * D + e * 256 + tid];
    s += v[e]; s2 += v[e] * v[e];
  }
#pragma unroll
  for (int o = 16; o > 0; o >>= 1) { s += __shfl_down(s, o, 32); s2 += __shfl_down(s2, o, 32); }
  __shared__ float w1[8], w2[8], smu, srv;
  if (lane == 0) { w1[wave] = s; w2[wave] = s2; }
  __syncthreads();
  if (tid == 0) {
    float a = 0.f, bb = 0.f;
#pragma unroll
    for (int i = 0; i < 8; ++i) { a += w1[i]; bb += w2[i]; }
    float mu = a / (float)D;
    smu = mu; srv = rsqrtf(bb / (float)D - mu * mu + 1e-5f);
  }
  __syncthreads();
#pragma unroll
  for (int e = 0; e < 3; ++e) {
    int c = e * 256 + tid;
    out[row * D + c] = (v[e] - smu) * srv * gamma[c] + beta[c];
  }
}

// =====================================================================
extern "C" void kernel_launch(void* const* d_in, const int* in_sizes, int n_in,
                              void* d_out, int out_size, void* d_ws, size_t ws_size,
                              hipStream_t stream) {
  const float*         x     = (const float*)d_in[0];
  const unsigned char* amask = (const unsigned char*)d_in[1];
  const float*         W_q   = (const float*)d_in[2];
  const float*         b_q   = (const float*)d_in[3];
  const float*         W_v   = (const float*)d_in[4];
  const float*         b_v   = (const float*)d_in[5];
  const float*         gamma = (const float*)d_in[6];
  const float*         beta  = (const float*)d_in[7];
  float*               out   = (float*)d_out;

  constexpr size_t QBYTES = (size_t)B * S * D * 2;      // bf16
  constexpr size_t CBYTES = (size_t)B * H * S * 4;      // f32
  char* ws = (char*)d_ws;
  unsigned short* qw      = (unsigned short*)ws;
  unsigned short* kvw     = (unsigned short*)(ws + QBYTES);
  float*          cmax    = (float*)(ws + 2 * QBYTES);
  float*          rcsum   = (float*)(ws + 2 * QBYTES + CBYTES);
  float*          attnb   = (float*)(ws + 2 * QBYTES + 2 * CBYTES);

  dim3 g1((B * S) / 64, D / 64);
  proj_kernel<<<g1, 128, 0, stream>>>(x, W_q, b_q, qw);
  proj_kernel<<<g1, 128, 0, stream>>>(x, W_v, b_v, kvw);

  dim3 g2(S / 128, B * H);
  colstats_kernel<<<g2, 256, 0, stream>>>(qw, kvw, amask, cmax, rcsum);

  dim3 g3(S / 32, B * H);
  attnout_kernel<<<g3, 192, 0, stream>>>(qw, kvw, amask, cmax, rcsum, x, attnb);

  ln_kernel<<<B * S, 256, 0, stream>>>(attnb, gamma, beta, out);
}
